// MAB_mk2_51986284151071
// MI455X (gfx1250) — matrix operand order repaired and checked by reading, not compiled, not run on hardware
//
#include <hip/hip_runtime.h>
#include <hip/hip_bf16.h>

#define D        512
#define NQ       1024
#define BATCH    16
#define M_ROWS   (BATCH * NQ)   // 16384
#define DH       64

typedef __attribute__((ext_vector_type(16))) __bf16 v16bf;
typedef __attribute__((ext_vector_type(8)))  __bf16 v8bf;
typedef __attribute__((ext_vector_type(8)))  float  v8f;
typedef int vi4 __attribute__((vector_size(16)));

#define WMMA_BF16(a, b, c) \
    __builtin_amdgcn_wmma_f32_16x16x32_bf16(false, (a), false, (b), (short)0, (c), false, false)

#if defined(__has_builtin)
#if __has_builtin(__builtin_amdgcn_global_load_async_to_lds_b128)
#define MAB_ASYNC_LDS 1
#endif
#endif

static __device__ __forceinline__ v8bf ld8bf(const __bf16* p) {
    return *(const v8bf*)p;
}
static __device__ __forceinline__ v16bf ld16bf(const __bf16* p) {
    return *(const v16bf*)p;
}
static __device__ __forceinline__ v16bf cat8(v8bf lo, v8bf hi) {
    v16bf r;
#pragma unroll
    for (int i = 0; i < 8; ++i) { r[i] = lo[i]; r[8 + i] = hi[i]; }
    return r;
}

// 16-byte global -> LDS copy: async path on gfx1250, sync fallback.
// Builtin signature (probe-confirmed by round-2 diagnostic):
//   void(v4i addrspace(1)* src, v4i addrspace(3)* dst, Imm int off, Imm int cpol)
static __device__ __forceinline__ void cp16_g2l(__bf16* lds, const __bf16* g) {
#ifdef MAB_ASYNC_LDS
    __builtin_amdgcn_global_load_async_to_lds_b128(
        (__attribute__((address_space(1))) vi4*)(g),
        (__attribute__((address_space(3))) vi4*)(lds), 0, 0);
#else
    *(v8bf*)lds = *(const v8bf*)g;
#endif
}
static __device__ __forceinline__ void wait_async_lds() {
#ifdef MAB_ASYNC_LDS
#if __has_builtin(__builtin_amdgcn_s_wait_asynccnt)
    __builtin_amdgcn_s_wait_asynccnt(0);
#else
    asm volatile("s_wait_asynccnt 0" ::: "memory");
#endif
#endif
}

// Build a 16x32 bf16 A-fragment row slice from f32 global (lane half kh owns
// K = kh*16..kh*16+15, contiguous float4 pairs).
static __device__ __forceinline__ v16bf afrag_from_f32(const float* row, int kb, int kh) {
    const float4* p0 = (const float4*)(row + kb + kh * 16);
    const float4* p1 = (const float4*)(row + kb + kh * 16 + 8);
    float4 a0 = p0[0], a1 = p0[1], a2 = p1[0], a3 = p1[1];
    v16bf af;
    af[0] = (__bf16)a0.x; af[1] = (__bf16)a0.y; af[2]  = (__bf16)a0.z; af[3]  = (__bf16)a0.w;
    af[4] = (__bf16)a1.x; af[5] = (__bf16)a1.y; af[6]  = (__bf16)a1.z; af[7]  = (__bf16)a1.w;
    af[8] = (__bf16)a2.x; af[9] = (__bf16)a2.y; af[10] = (__bf16)a2.z; af[11] = (__bf16)a2.w;
    af[12] = (__bf16)a3.x; af[13] = (__bf16)a3.y; af[14] = (__bf16)a3.z; af[15] = (__bf16)a3.w;
    return af;
}

// ---------------------------------------------------------------------------
// Transpose + convert a 512x512 f32 weight into bf16 [out_col, in_k].
// ---------------------------------------------------------------------------
__global__ __launch_bounds__(256) void mab_wtrans_kernel(const float* __restrict__ W,
                                                         __bf16* __restrict__ WT) {
    int idx = blockIdx.x * 256 + threadIdx.x;
    int n = idx >> 9;
    int k = idx & 511;
    WT[(size_t)n * D + k] = (__bf16)W[(size_t)k * D + n];
}

// ---------------------------------------------------------------------------
// Projection GEMM: out = (A[M,512] @ W + bias) [* mask]  (bf16 out)
// Block = 256 threads = 8 waves (2 row groups x 4 col strips).
// Each wave: 32x64 strip (2 M-tiles share every B-fragment -> 8 WMMA/K-step).
// ---------------------------------------------------------------------------
template <bool HAS_MASK, bool TRANS_OUT>
__global__ __launch_bounds__(256) void mab_proj_gemm_kernel(
    const float* __restrict__ A, const __bf16* __restrict__ WT,
    const float* __restrict__ bias, const float* __restrict__ mask,
    __bf16* __restrict__ outp) {
    const int lane = threadIdx.x & 31;
    const int wave = threadIdx.x >> 5;
    const int wx = wave & 3;
    const int wy = wave >> 2;
    const int m  = lane & 15;
    const int kh = lane >> 4;
    const int mbase = blockIdx.y * 64 + wy * 32;
    const int nbase = blockIdx.x * 256 + wx * 64;

    const float* Arow0 = A + (size_t)(mbase + m) * D;
    const float* Arow1 = A + (size_t)(mbase + 16 + m) * D;

    v8f acc[2][4] = {};
    for (int kb = 0; kb < D; kb += 32) {
        __builtin_prefetch(Arow0 + kb + 64, 0, 0);
        __builtin_prefetch(Arow1 + kb + 64, 0, 0);
        v16bf af0 = afrag_from_f32(Arow0, kb, kh);
        v16bf af1 = afrag_from_f32(Arow1, kb, kh);
#pragma unroll
        for (int t = 0; t < 4; ++t) {
            int n = nbase + t * 16 + m;
            v16bf bf = ld16bf(WT + (size_t)n * D + kb + kh * 16);
            acc[0][t] = WMMA_BF16(af0, bf, acc[0][t]);
            acc[1][t] = WMMA_BF16(af1, bf, acc[1][t]);
        }
    }

#pragma unroll
    for (int r2 = 0; r2 < 2; ++r2) {
        float mv[8];
#pragma unroll
        for (int j = 0; j < 8; ++j) {
            int row = mbase + r2 * 16 + j + 8 * kh;
            mv[j] = HAS_MASK ? mask[row] : 1.0f;
        }
#pragma unroll
        for (int t = 0; t < 4; ++t) {
            int col = nbase + t * 16 + m;
            float bb = bias[col];
#pragma unroll
            for (int j = 0; j < 8; ++j) {
                int row = mbase + r2 * 16 + j + 8 * kh;   // D-layout M = j + 8*kh
                float v = (acc[r2][t][j] + bb) * mv[j];
                __bf16 bv = (__bf16)v;
                if (TRANS_OUT) {
                    int bidx = row >> 10, key = row & (NQ - 1);
                    outp[((size_t)bidx * D + col) * NQ + key] = bv;
                } else {
                    outp[(size_t)row * D + col] = bv;
                }
            }
        }
    }
}

// ---------------------------------------------------------------------------
// Attention: block = 8 waves = 128 queries of one (batch, head).
// K/V tiles (32 keys) staged once per block in LDS via async global->LDS,
// double-buffered; all 8 waves consume them (8x global traffic reduction).
// X = exp(exp(S/sqrt(512))*Km)*Qm re-striped per wave through LDS to feed the
// A*V WMMAs; residual Qh and 1/(den+1e-8) applied at the end.
// ---------------------------------------------------------------------------
__global__ __launch_bounds__(256) void mab_attn_kernel(
    const __bf16* __restrict__ Qp, const __bf16* __restrict__ Kp,
    const __bf16* __restrict__ Vt, const float* __restrict__ Qm,
    const float* __restrict__ Km, float* __restrict__ O) {
    __shared__ alignas(16) __bf16 Klds[2][32 * 64];   // [key][feat]
    __shared__ alignas(16) __bf16 Vlds[2][64 * 32];   // [feat][key]
    __shared__ alignas(16) __bf16 xlds[8][16 * 32];
    __shared__ float denlds[8][32];

    const int tid  = threadIdx.x;
    const int lane = tid & 31;
    const int wave = tid >> 5;
    const int m  = lane & 15;
    const int kh = lane >> 4;
    const int b = blockIdx.z, head = blockIdx.y;
    const int hb = head * DH;
    const int q0 = blockIdx.x * 128 + wave * 16;
    const float sc = 0.044194173824159216f;           // 1/sqrt(512)

    const __bf16* Kb  = Kp + (size_t)b * NQ * D;
    const __bf16* Vb  = Vt + (size_t)b * D * NQ;
    const float*  Kmb = Km + b * NQ;

    // Per-thread staging slots: K tile = 32 rows x 8 chunks of 16B,
    //                           V tile = 64 rows x 4 chunks of 16B.
    const int kKey = tid >> 3, kSub = tid & 7;
    const int vFt  = tid >> 2, vSub = tid & 3;

    // Q A-fragments (dh = 64 -> two K=32 WMMA steps), loaded once.
    const size_t rowQ = (size_t)(b * NQ + q0 + m) * D + hb;
    v16bf afQ0 = cat8(ld8bf(Qp + rowQ + kh * 16), ld8bf(Qp + rowQ + 8 + kh * 16));
    v16bf afQ1 = cat8(ld8bf(Qp + rowQ + 32 + kh * 16), ld8bf(Qp + rowQ + 40 + kh * 16));

    float qmv[8];
#pragma unroll
    for (int j = 0; j < 8; ++j)
        qmv[j] = Qm[b * NQ + q0 + j + 8 * kh];

    v8f acc[4] = {};
    float denp = 0.0f;
    __bf16* xw = xlds[wave];

    // Preload first K/V tile (kb = 0) into buffer 0.
    cp16_g2l(&Klds[0][kKey * 64 + kSub * 8], Kb + (size_t)kKey * D + hb + kSub * 8);
    cp16_g2l(&Vlds[0][vFt * 32 + vSub * 8], Vb + (size_t)(hb + vFt) * NQ + vSub * 8);

    for (int kb = 0; kb < NQ; kb += 32) {
        const int cur = (kb >> 5) & 1;
        wait_async_lds();
        __syncthreads();                   // tile[cur] ready; tile[!cur] free
        if (kb + 32 < NQ) {
            const int nkb = kb + 32;
            cp16_g2l(&Klds[cur ^ 1][kKey * 64 + kSub * 8],
                     Kb + (size_t)(nkb + kKey) * D + hb + kSub * 8);
            cp16_g2l(&Vlds[cur ^ 1][vFt * 32 + vSub * 8],
                     Vb + (size_t)(hb + vFt) * NQ + nkb + vSub * 8);
        }

        // S tiles from the staged K tile (keys [kb..kb+15], [kb+16..kb+31]).
        const __bf16* Kc = Klds[cur];
        v8f s0 = {}, s1 = {};
        s0 = WMMA_BF16(afQ0, ld16bf(Kc + m * 64 + kh * 16), s0);
        s0 = WMMA_BF16(afQ1, ld16bf(Kc + m * 64 + 32 + kh * 16), s0);
        s1 = WMMA_BF16(afQ0, ld16bf(Kc + (16 + m) * 64 + kh * 16), s1);
        s1 = WMMA_BF16(afQ1, ld16bf(Kc + (16 + m) * 64 + 32 + kh * 16), s1);

        float km0 = Kmb[kb + m];
        float km1 = Kmb[kb + 16 + m];
#pragma unroll
        for (int j = 0; j < 8; ++j) {
            int row = j + 8 * kh;
            float x0 = __expf(__expf(s0[j] * sc) * km0) * qmv[j];
            float x1 = __expf(__expf(s1[j] * sc) * km1) * qmv[j];
            xw[row * 32 + m]      = (__bf16)x0;
            xw[row * 32 + 16 + m] = (__bf16)x1;
        }
        // LDS in-order within a wave: read back as 16x32 A-fragment.
        v16bf afX = cat8(*(const v8bf*)(xw + m * 32 + kh * 16),
                         *(const v8bf*)(xw + m * 32 + 8 + kh * 16));
#pragma unroll
        for (int i = 0; i < 16; ++i) denp += (float)afX[i];

        const __bf16* Vc = Vlds[cur];
#pragma unroll
        for (int t = 0; t < 4; ++t)
            acc[t] = WMMA_BF16(afX, ld16bf(Vc + (t * 16 + m) * 32 + kh * 16), acc[t]);
    }

    denlds[wave][lane] = denp;
    float invden[8];
#pragma unroll
    for (int j = 0; j < 8; ++j) {
        int r = j + 8 * kh;
        invden[j] = 1.0f / (denlds[wave][r] + denlds[wave][r + 16] + 1e-8f);
    }

#pragma unroll
    for (int t = 0; t < 4; ++t) {
        int col = hb + t * 16 + m;
#pragma unroll
        for (int j = 0; j < 8; ++j) {
            int rowg = b * NQ + q0 + j + 8 * kh;
            float resid = (float)Qp[(size_t)rowg * D + col];
            O[(size_t)rowg * D + col] = resid + acc[t][j] * invden[j];
        }
    }
}

// ---------------------------------------------------------------------------
// Rowwise LayerNorm over 512 features; writes f32 in place + bf16 copy.
// ---------------------------------------------------------------------------
__global__ __launch_bounds__(256) void mab_ln_kernel(float* __restrict__ X,
                                                     __bf16* __restrict__ Xb,
                                                     const float* __restrict__ g,
                                                     const float* __restrict__ bta) {
    __shared__ float ssum[256];
    __shared__ float ssq[256];
    const int row = blockIdx.x, tid = threadIdx.x;
    float* xr = X + (size_t)row * D;
    float v0 = xr[tid], v1 = xr[tid + 256];
    ssum[tid] = v0 + v1;
    ssq[tid]  = v0 * v0 + v1 * v1;
    __syncthreads();
    for (int s = 128; s > 0; s >>= 1) {
        if (tid < s) { ssum[tid] += ssum[tid + s]; ssq[tid] += ssq[tid + s]; }
        __syncthreads();
    }
    float mu  = ssum[0] * (1.0f / D);
    float var = ssq[0] * (1.0f / D) - mu * mu;
    float rs  = rsqrtf(var + 1e-5f);
    float y0 = (v0 - mu) * rs * g[tid] + bta[tid];
    float y1 = (v1 - mu) * rs * g[tid + 256] + bta[tid + 256];
    xr[tid] = y0;
    xr[tid + 256] = y1;
    Xb[(size_t)row * D + tid]       = (__bf16)y0;
    Xb[(size_t)row * D + tid + 256] = (__bf16)y1;
}

// ---------------------------------------------------------------------------
// Fused: Y = LN1(X + relu(X @ Wo + bo)). One block = 16 rows x 512 cols.
// ---------------------------------------------------------------------------
__global__ __launch_bounds__(256) void mab_out_gemm_ln_kernel(
    const __bf16* __restrict__ Xb, const float* __restrict__ Xf,
    const __bf16* __restrict__ WoT, const float* __restrict__ bo,
    const float* __restrict__ g1, const float* __restrict__ b1,
    float* __restrict__ out) {
    __shared__ float tile[16 * D];       // 32 KB
    __shared__ float red[16][16][2];
    __shared__ float stats[16][2];

    const int lane = threadIdx.x & 31;
    const int wave = threadIdx.x >> 5;
    const int m  = lane & 15;
    const int kh = lane >> 4;
    const int rbase = blockIdx.x * 16;
    const int nb = wave * 64;

    v8f acc[4] = {};
    const __bf16* Arow = Xb + (size_t)(rbase + m) * D;
    for (int kb = 0; kb < D; kb += 32) {
        v16bf af = cat8(ld8bf(Arow + kb + kh * 16), ld8bf(Arow + kb + 8 + kh * 16));
#pragma unroll
        for (int t = 0; t < 4; ++t) {
            int n = nb + t * 16 + m;
            acc[t] = WMMA_BF16(af, ld16bf(WoT + (size_t)n * D + kb + kh * 16), acc[t]);
        }
    }

#pragma unroll
    for (int t = 0; t < 4; ++t) {
        int c = nb + t * 16 + m;
        float bb = bo[c];
#pragma unroll
        for (int j = 0; j < 8; ++j) {
            int r = j + 8 * kh;
            float v = fmaxf(acc[t][j] + bb, 0.0f) + Xf[(size_t)(rbase + r) * D + c];
            tile[r * D + c] = v;
        }
    }
    __syncthreads();

    const int rr = threadIdx.x >> 4;
    const int cc = threadIdx.x & 15;
    float s = 0.0f, q = 0.0f;
#pragma unroll
    for (int i = 0; i < 32; ++i) {
        float v = tile[rr * D + cc * 32 + i];
        s += v; q += v * v;
    }
    red[rr][cc][0] = s;
    red[rr][cc][1] = q;
    __syncthreads();
    if (threadIdx.x < 16) {
        int r = threadIdx.x;
        float S = 0.0f, Q2 = 0.0f;
        for (int i = 0; i < 16; ++i) { S += red[r][i][0]; Q2 += red[r][i][1]; }
        float mu  = S * (1.0f / D);
        float var = Q2 * (1.0f / D) - mu * mu;
        stats[r][0] = mu;
        stats[r][1] = rsqrtf(var + 1e-5f);
    }
    __syncthreads();
#pragma unroll
    for (int i = 0; i < 32; ++i) {
        int c = cc * 32 + i;
        float v = tile[rr * D + c];
        out[(size_t)(rbase + rr) * D + c] = (v - stats[rr][0]) * stats[rr][1] * g1[c] + b1[c];
    }
}

// ---------------------------------------------------------------------------
extern "C" void kernel_launch(void* const* d_in, const int* in_sizes, int n_in,
                              void* d_out, int out_size, void* d_ws, size_t ws_size,
                              hipStream_t stream) {
    const float* Q  = (const float*)d_in[0];
    const float* K  = (const float*)d_in[1];
    const float* Qm = (const float*)d_in[2];
    const float* Km = (const float*)d_in[3];
    const float* Wq = (const float*)d_in[4];
    const float* bq = (const float*)d_in[5];
    const float* Wk = (const float*)d_in[6];
    const float* bk = (const float*)d_in[7];
    const float* Wv = (const float*)d_in[8];
    const float* bv = (const float*)d_in[9];
    const float* Wo = (const float*)d_in[10];
    const float* bo = (const float*)d_in[11];
    const float* g0 = (const float*)d_in[12];
    const float* b0 = (const float*)d_in[13];
    const float* g1 = (const float*)d_in[14];
    const float* b1 = (const float*)d_in[15];

    char* ws = (char*)d_ws;
    size_t o = 0;
    __bf16* WqT = (__bf16*)(ws + o); o += (size_t)D * D * 2;
    __bf16* WkT = (__bf16*)(ws + o); o += (size_t)D * D * 2;
    __bf16* WvT = (__bf16*)(ws + o); o += (size_t)D * D * 2;
    __bf16* WoT = (__bf16*)(ws + o); o += (size_t)D * D * 2;
    __bf16* Qp  = (__bf16*)(ws + o); o += (size_t)M_ROWS * D * 2;     // masked Q proj
    __bf16* Kp  = (__bf16*)(ws + o); o += (size_t)M_ROWS * D * 2;     // masked K proj
    __bf16* Vt  = (__bf16*)(ws + o); o += (size_t)BATCH * D * NQ * 2; // V transposed
    float*  AO  = (float*)(ws + o);  o += (size_t)M_ROWS * D * 4;     // attn out / LN0 out
    __bf16* X0b = (__bf16*)(ws + o); o += (size_t)M_ROWS * D * 2;     // LN0 out (bf16)

    // 1) Weight transpose + bf16 convert
    mab_wtrans_kernel<<<1024, 256, 0, stream>>>(Wq, WqT);
    mab_wtrans_kernel<<<1024, 256, 0, stream>>>(Wk, WkT);
    mab_wtrans_kernel<<<1024, 256, 0, stream>>>(Wv, WvT);
    mab_wtrans_kernel<<<1024, 256, 0, stream>>>(Wo, WoT);

    // 2) Projections (bias + feature masking fused); V written transposed
    dim3 pg(2, M_ROWS / 64);
    mab_proj_gemm_kernel<true,  false><<<pg, 256, 0, stream>>>(Q, WqT, bq, Qm, Qp);
    mab_proj_gemm_kernel<true,  false><<<pg, 256, 0, stream>>>(K, WkT, bk, Km, Kp);
    mab_proj_gemm_kernel<false, true ><<<pg, 256, 0, stream>>>(K, WvT, bv, nullptr, Vt);

    // 3) Attention (double-exp masked softmax, residual fused, async K/V staging)
    mab_attn_kernel<<<dim3(NQ / 128, 8, BATCH), 256, 0, stream>>>(Qp, Kp, Vt, Qm, Km, AO);

    // 4) LayerNorm 0 (in place + bf16 copy)
    mab_ln_kernel<<<M_ROWS, 256, 0, stream>>>(AO, X0b, g0, b0);

    // 5) Fused Wo GEMM + relu + residual + LayerNorm 1 -> output
    mab_out_gemm_ln_kernel<<<M_ROWS / 16, 256, 0, stream>>>(X0b, AO, WoT, bo, g1, b1,
                                                            (float*)d_out);
}
